// BiCEBertAttention_49495203119445
// MI455X (gfx1250) — matrix operand order repaired and checked by reading, not compiled, not run on hardware
//
#include <hip/hip_runtime.h>
#include <hip/hip_bf16.h>
#include <cstddef>

typedef _Float16 h16;
typedef __attribute__((ext_vector_type(16))) _Float16 v16h;
typedef __attribute__((ext_vector_type(8)))  _Float16 v8h;
typedef __attribute__((ext_vector_type(8)))  float    v8f;
typedef __attribute__((ext_vector_type(4)))  float    v4f;

#define DEV static __device__ __forceinline__

// ---- problem constants ----
#define B_   2
#define T_   2048
#define C_   768
#define H_   12
#define D_   64
#define NL_  6      // N_LEFT (causal heads)

// ---------------- fragment helpers ----------------

DEV v16h pack2(v8h lo, v8h hi) {
  v16h r;
#pragma unroll
  for (int i = 0; i < 8; ++i) { r[i] = lo[i]; r[i + 8] = hi[i]; }
  return r;
}

DEV v8h ldg8h(const h16* p) { return *(const v8h*)p; }

DEV v8h cvt8(const float* p) {
  v4f a = *(const v4f*)p;
  v4f b = *(const v4f*)(p + 4);
  v8h r;
#pragma unroll
  for (int i = 0; i < 4; ++i) { r[i] = (h16)a[i]; r[i + 4] = (h16)b[i]; }
  return r;
}

DEV v8f wmma_f16(v16h a, v16h b, v8f c) {
  return __builtin_amdgcn_wmma_f32_16x16x32_f16(
      false, a, false, b, (short)0, c, false, false);
}

// row-wise reductions: a C-matrix row's 16 elements live in one 16-lane
// half-group, so xor masks 1,2,4,8 suffice.
DEV float rmax16(float v) {
#pragma unroll
  for (int o = 1; o < 16; o <<= 1) v = fmaxf(v, __shfl_xor(v, o, 32));
  return v;
}
DEV float rsum16(float v) {
#pragma unroll
  for (int o = 1; o < 16; o <<= 1) v += __shfl_xor(v, o, 32);
  return v;
}

// ---------------- kernel 1: QKV projection ----------------
// out[t, f] = sum_c x[t,c] * Wqkv[f,c] + b[f]
// grid = (token tiles, three, feature-tile groups): blockIdx.y selects Q/K/V
// with a *scalar* branch (no per-lane divide, no exec-mask divergence).
// Q,K -> [B,H,T,D] f16 ; V -> [B,H,D,T] f16 (transposed for the PV GEMM)
__global__ __launch_bounds__(128)
void bice_qkv_kernel(const float* __restrict__ x,
                     const float* __restrict__ W,
                     const float* __restrict__ bias,
                     h16* __restrict__ Qb, h16* __restrict__ Kb,
                     h16* __restrict__ Vt) {
  const int lane = threadIdx.x & 31;
  const int wave = threadIdx.x >> 5;
  const int lm = lane & 15, hf = lane >> 4;

  const int t0    = blockIdx.x * 16;                 // token tile
  const int three = blockIdx.y;                      // SGPR: 0=Q 1=K 2=V
  const int f_in  = (blockIdx.z * 4 + wave) * 16;    // feature within chunk
  const int f     = three * C_ + f_in;               // full qkv row

  const float* arow = x + (size_t)(t0 + lm) * C_;    // A: row = token
  const float* brow = W + (size_t)(f + lm) * C_;     // B: col = feature

  v8f c;
#pragma unroll
  for (int i = 0; i < 8; ++i) c[i] = 0.0f;

  for (int k = 0; k < C_; k += 32) {
    v16h a = pack2(cvt8(arow + k + hf * 16), cvt8(arow + k + hf * 16 + 8));
    v16h b = pack2(cvt8(brow + k + hf * 16), cvt8(brow + k + hf * 16 + 8));
    c = wmma_f16(a, b, c);
  }

  const int h = f_in >> 6;              // head (shift, no divide)
  const int d = (f_in & 48) + lm;       // dim within head
  const float bv = bias[f + lm];

  const int bidx = t0 >> 11;            // tile never crosses batch boundary
  const int tloc = (t0 & (T_ - 1)) + hf * 8;
  const size_t bh = (size_t)bidx * H_ + h;

  if (three == 0) {                     // scalar s_cbranch on blockIdx.y
    h16* p = Qb + (bh * T_ + tloc) * D_ + d;
#pragma unroll
    for (int r = 0; r < 8; ++r) p[(size_t)r * D_] = (h16)(c[r] + bv);
  } else if (three == 1) {
    h16* p = Kb + (bh * T_ + tloc) * D_ + d;
#pragma unroll
    for (int r = 0; r < 8; ++r) p[(size_t)r * D_] = (h16)(c[r] + bv);
  } else {
    h16* p = Vt + (bh * D_ + d) * T_ + tloc;
#pragma unroll
    for (int r = 0; r < 8; ++r) p[r] = (h16)(c[r] + bv);
  }
}

// ---------------- kernel 2: flash attention ----------------
// One wave per 16-query tile; 32-key blocks with online softmax.
// heads [0,NL): causal (mask k > q); heads [NL,H): anti-causal (mask k < q).

template <bool LEFT, bool MASKED>
DEV void attn_step(int k0, int q0, int lm, int hf,
                   const v16h& qf0, const v16h& qf1,
                   const h16* __restrict__ Kbase,
                   const h16* __restrict__ Vbase,
                   const int* __restrict__ amrow,
                   h16* __restrict__ P,
                   float* mrow, float* lrow, v8f* acc,
                   bool prefetch) {
  const int kA = k0 + lm;        // key col, S-tile 0
  const int kB = k0 + 16 + lm;   // key col, S-tile 1

  // B-fragments of K^T (col = key row, 16 contiguous d per lane)
  const h16* kp0 = Kbase + (size_t)kA * D_;
  const h16* kp1 = Kbase + (size_t)kB * D_;
  const v16h kb00 = pack2(ldg8h(kp0 + hf * 16), ldg8h(kp0 + hf * 16 + 8));
  const v16h kb01 = pack2(ldg8h(kp0 + 32 + hf * 16), ldg8h(kp0 + 32 + hf * 16 + 8));
  const v16h kb10 = pack2(ldg8h(kp1 + hf * 16), ldg8h(kp1 + hf * 16 + 8));
  const v16h kb11 = pack2(ldg8h(kp1 + 32 + hf * 16), ldg8h(kp1 + 32 + hf * 16 + 8));

  if (prefetch) {  // warm next block's K rows in WGP$ (global_prefetch_b8)
    __builtin_prefetch(Kbase + (size_t)(kA + 32) * D_, 0, 1);
    __builtin_prefetch(Kbase + (size_t)(kB + 32) * D_, 0, 1);
  }

  v8f s0, s1;
#pragma unroll
  for (int i = 0; i < 8; ++i) { s0[i] = 0.0f; s1[i] = 0.0f; }
  s0 = wmma_f16(qf0, kb00, s0);
  s0 = wmma_f16(qf1, kb01, s0);
  s1 = wmma_f16(qf0, kb10, s1);
  s1 = wmma_f16(qf1, kb11, s1);

  const int am0 = amrow[kA];
  const int am1 = amrow[kB];

  const float SC  = 0.18033688011112042f;  // D^-1/2 * log2(e)
  const float NEG = -1e30f;

#pragma unroll
  for (int r = 0; r < 8; ++r) {
    float v0 = s0[r] * SC;
    float v1 = s1[r] * SC;
    if (!am0) v0 = NEG;
    if (!am1) v1 = NEG;
    if (MASKED) {
      const int q = q0 + hf * 8 + r;
      if (LEFT) { if (kA > q) v0 = NEG; if (kB > q) v1 = NEG; }
      else      { if (kA < q) v0 = NEG; if (kB < q) v1 = NEG; }
    }
    const float mx = rmax16(fmaxf(v0, v1));
    const float mn = fmaxf(mrow[r], mx);
    const float al = exp2f(mrow[r] - mn);
    const float p0 = exp2f(v0 - mn);
    const float p1 = exp2f(v1 - mn);
    lrow[r] = lrow[r] * al + rsum16(p0 + p1);
    mrow[r] = mn;
#pragma unroll
    for (int nt = 0; nt < 4; ++nt) acc[nt][r] *= al;
    // stage P in D-layout (row = q, col = key within block)
    P[(hf * 8 + r) * 32 + lm]      = (h16)p0;
    P[(hf * 8 + r) * 32 + 16 + lm] = (h16)p1;
  }

  // re-stripe: read P back as an A-fragment (row = lm, K = 32 keys)
  const v16h pf = pack2(*(const v8h*)&P[lm * 32 + hf * 16],
                        *(const v8h*)&P[lm * 32 + hf * 16 + 8]);

#pragma unroll
  for (int nt = 0; nt < 4; ++nt) {
    const h16* vp = Vbase + (size_t)(nt * 16 + lm) * T_ + k0;
    const v16h vb = pack2(ldg8h(vp + hf * 16), ldg8h(vp + hf * 16 + 8));
    acc[nt] = wmma_f16(pf, vb, acc[nt]);
  }
}

__global__ __launch_bounds__(128)
void bice_attn_kernel(const h16* __restrict__ Qb, const h16* __restrict__ Kb,
                      const h16* __restrict__ Vt, const int* __restrict__ amask,
                      h16* __restrict__ ctx) {
  __shared__ alignas(16) h16 Pbuf[4][16 * 32];

  const int lane = threadIdx.x & 31;
  const int wave = threadIdx.x >> 5;
  const int lm = lane & 15, hf = lane >> 4;

  const int bh = blockIdx.y;
  const int b = bh / H_;                 // scalar (blockIdx.y is SGPR)
  const int h = bh - b * H_;
  const int q0 = (blockIdx.x * 4 + wave) * 16;
  const bool left = (h < NL_);

  const h16* Qp = Qb + ((size_t)bh * T_ + q0) * D_;
  const v16h qf0 = pack2(ldg8h(Qp + lm * D_ + hf * 16),
                         ldg8h(Qp + lm * D_ + hf * 16 + 8));
  const v16h qf1 = pack2(ldg8h(Qp + lm * D_ + 32 + hf * 16),
                         ldg8h(Qp + lm * D_ + 32 + hf * 16 + 8));

  float mrow[8], lrow[8];
  v8f acc[4];
#pragma unroll
  for (int r = 0; r < 8; ++r) { mrow[r] = -1e30f; lrow[r] = 0.0f; }
#pragma unroll
  for (int nt = 0; nt < 4; ++nt)
#pragma unroll
    for (int r = 0; r < 8; ++r) acc[nt][r] = 0.0f;

  const h16* Kbase = Kb + (size_t)bh * T_ * D_;
  const h16* Vbase = Vt + (size_t)bh * D_ * T_;
  const int* amrow = amask + b * T_;
  h16* P = Pbuf[wave];

  if (left) {
    // interior: blocks entirely at-or-below the diagonal (k0+31 <= q0)
    const int IL = ((q0 + 1) >> 5) << 5;
    for (int k0 = 0; k0 < IL; k0 += 32)
      attn_step<true, false>(k0, q0, lm, hf, qf0, qf1, Kbase, Vbase, amrow, P,
                             mrow, lrow, acc, k0 + 32 < IL);
    // diagonal region
    for (int k0 = IL; k0 < q0 + 16; k0 += 32)
      attn_step<true, true>(k0, q0, lm, hf, qf0, qf1, Kbase, Vbase, amrow, P,
                            mrow, lrow, acc, false);
  } else {
    int lo = q0 - 31; if (lo < 0) lo = 0;
    const int k0s = (lo + 31) & ~31;
    int IR = ((q0 + 15 + 31) >> 5) << 5;   // first block with k0 >= q0+15
    if (IR > T_) IR = T_;
    // diagonal region first
    for (int k0 = k0s; k0 < IR; k0 += 32)
      attn_step<false, true>(k0, q0, lm, hf, qf0, qf1, Kbase, Vbase, amrow, P,
                             mrow, lrow, acc, false);
    // interior: blocks entirely at-or-above the diagonal
    for (int k0 = IR; k0 < T_; k0 += 32)
      attn_step<false, false>(k0, q0, lm, hf, qf0, qf1, Kbase, Vbase, amrow, P,
                              mrow, lrow, acc, k0 + 32 < T_);
  }

  // epilogue: O /= l, write ctx [B,T,C] f16
#pragma unroll
  for (int r = 0; r < 8; ++r) {
    const float inv = 1.0f / lrow[r];
    const size_t row = (size_t)(b * T_ + q0 + hf * 8 + r) * C_ + h * D_;
#pragma unroll
    for (int nt = 0; nt < 4; ++nt)
      ctx[row + nt * 16 + lm] = (h16)(acc[nt][r] * inv);
  }
}

// ---------------- kernel 3: output projection ----------------
__global__ __launch_bounds__(128)
void bice_out_kernel(const h16* __restrict__ ctx,
                     const float* __restrict__ W,
                     const float* __restrict__ bias,
                     float* __restrict__ out) {
  const int lane = threadIdx.x & 31;
  const int wave = threadIdx.x >> 5;
  const int lm = lane & 15, hf = lane >> 4;

  const int t0 = blockIdx.x * 16;
  const int f0 = (blockIdx.y * 4 + wave) * 16;

  const h16* arow = ctx + (size_t)(t0 + lm) * C_;
  const float* brow = W + (size_t)(f0 + lm) * C_;

  v8f c;
#pragma unroll
  for (int i = 0; i < 8; ++i) c[i] = 0.0f;

  for (int k = 0; k < C_; k += 32) {
    v16h a = pack2(ldg8h(arow + k + hf * 16), ldg8h(arow + k + hf * 16 + 8));
    v16h b = pack2(cvt8(brow + k + hf * 16), cvt8(brow + k + hf * 16 + 8));
    c = wmma_f16(a, b, c);
  }

  const float bv = bias[f0 + lm];
  float* orow = out + (size_t)(t0 + hf * 8) * C_ + f0 + lm;
#pragma unroll
  for (int r = 0; r < 8; ++r) orow[(size_t)r * C_] = c[r] + bv;
}

// ---------------- launch ----------------
extern "C" void kernel_launch(void* const* d_in, const int* in_sizes, int n_in,
                              void* d_out, int out_size, void* d_ws, size_t ws_size,
                              hipStream_t stream) {
  const float* x      = (const float*)d_in[0];
  const int*   amask  = (const int*)d_in[1];
  const float* Wqkv_w = (const float*)d_in[2];
  const float* Wqkv_b = (const float*)d_in[3];
  const float* Wo_w   = (const float*)d_in[4];
  const float* Wo_b   = (const float*)d_in[5];

  const size_t nPer = (size_t)B_ * H_ * T_ * D_;  // 3,145,728 elems
  h16* Qb  = (h16*)d_ws;
  h16* Kb  = Qb + nPer;
  h16* Vt  = Kb + nPer;
  h16* ctx = Vt + nPer;                            // total 24 MiB of ws

  // QKV: 256 token tiles x {Q,K,V} x 12 blocks of 4 feature tiles
  bice_qkv_kernel<<<dim3((B_ * T_) / 16, 3, C_ / 64), 128, 0, stream>>>(
      x, Wqkv_w, Wqkv_b, Qb, Kb, Vt);

  // attention: 32 q-tile groups x (B*H) heads
  bice_attn_kernel<<<dim3(T_ / 64, B_ * H_), 128, 0, stream>>>(
      Qb, Kb, Vt, amask, ctx);

  // output projection: 256 token tiles x 12 blocks of 4 feature tiles
  bice_out_kernel<<<dim3((B_ * T_) / 16, C_ / 64), 128, 0, stream>>>(
      ctx, Wo_w, Wo_b, (float*)d_out);
}
